// DECSeq6_41180146434799
// MI455X (gfx1250) — matrix operand order repaired and checked by reading, not compiled, not run on hardware
//
#include <hip/hip_runtime.h>
#include <hip/hip_bf16.h>

// ---------------------------------------------------------------------------
// DGCNN-ish forward for MI455X (gfx1250, wave32, WMMA).
// All MLP GEMMs run through v_wmma_f32_16x16x32_f16 with f32 accumulation and
// a fused bias+ReLU+BatchNorm epilogue. Weights are pre-packed into the WMMA
// B-fragment layout so every fragment load is a contiguous global_load_b128.
// Wave tile is 32x64 (two A fragments share each B fragment) so the inner
// loop issues 12 b128 loads per 8 WMMAs.
// ---------------------------------------------------------------------------

typedef __attribute__((ext_vector_type(16))) _Float16 v16h;
typedef __attribute__((ext_vector_type(8)))  _Float16 v8h;
typedef __attribute__((ext_vector_type(8)))  float    v8f;

#define BN_EPS 1e-5f

__device__ __forceinline__ v16h load_frag16(const _Float16* p0, const _Float16* p1) {
    v8h lo = *(const v8h*)p0;
    v8h hi = *(const v8h*)p1;
    return __builtin_shufflevector(lo, hi,
        0, 1, 2, 3, 4, 5, 6, 7, 8, 9, 10, 11, 12, 13, 14, 15);
}

// ---------------------------------------------------------------------------
// Weight packing: W[K,N] f32 (row-major) -> Bp[K/32][N][32] f16.
// For K-step kt, column n, the 32 halves Bp[(kt*N+n)*32 + 0..31] hold
// W[kt*32+kk][n].  A lane's B fragment (16 halves, K lo/hi split by lane>=16)
// is then two contiguous b128 loads.
// ---------------------------------------------------------------------------
__global__ void pack_w_kernel(const float* __restrict__ W,
                              _Float16* __restrict__ Bp, int K, int N) {
    int t = blockIdx.x * blockDim.x + threadIdx.x;
    if (t >= K * N) return;
    int k = t / N, n = t - k * N;
    Bp[(((size_t)(k >> 5)) * N + n) * 32 + (k & 31)] = (_Float16)W[t];
}

// ---------------------------------------------------------------------------
// conv0: per-point Linear(3,32)+ReLU -> xf f32 [P,32]
// ---------------------------------------------------------------------------
__global__ void conv0_kernel(const float* __restrict__ pos,
                             const float* __restrict__ w0,
                             const float* __restrict__ b0,
                             float* __restrict__ xf, int P) {
    int p = blockIdx.x * blockDim.x + threadIdx.x;
    if (p >= P) return;
    float p0 = pos[3 * p], p1 = pos[3 * p + 1], p2 = pos[3 * p + 2];
    float* o = xf + (size_t)p * 32;
#pragma unroll
    for (int c = 0; c < 32; ++c) {
        float z = b0[c] + p0 * w0[c] + p1 * w0[32 + c] + p2 * w0[64 + c];
        o[c] = z > 0.f ? z : 0.f;
    }
}

// ---------------------------------------------------------------------------
// Fused kNN (k=5, self included, stable ties like lax.top_k) + gather +
// edge-feature build [x_i, x_j - x_i] written as f16.
// One 32-thread block per sample; sample features staged in LDS.
// ---------------------------------------------------------------------------
template <int C>
__global__ __launch_bounds__(32)
void knn_edge_kernel(const float* __restrict__ x,   // [B*32, C] f32
                     _Float16* __restrict__ h) {    // [B*32*5, 2C] f16
    __shared__ float xs[32][C];
    const int s = blockIdx.x;
    const int i = threadIdx.x;
    const float* xp = x + ((size_t)s * 32 + i) * C;
    for (int c = 0; c < C; ++c) xs[i][c] = xp[c];
    __syncthreads();

    float bd[5];
    int   bj[5];
#pragma unroll
    for (int q = 0; q < 5; ++q) { bd[q] = 3.4e38f; bj[q] = 0; }
    for (int j = 0; j < 32; ++j) {
        float d = 0.f;
        for (int c = 0; c < C; ++c) {
            float t = xs[i][c] - xs[j][c];
            d += t * t;
        }
        if (d < bd[4]) {                 // stable insertion (strict <):
            int p = 4;                   // equal distances keep lower index
            while (p > 0 && d < bd[p - 1]) {
                bd[p] = bd[p - 1]; bj[p] = bj[p - 1]; --p;
            }
            bd[p] = d; bj[p] = j;
        }
    }
    _Float16* hp = h + ((size_t)(s * 32 + i) * 5) * (2 * C);
#pragma unroll 1
    for (int q = 0; q < 5; ++q) {
        int j = bj[q];
        for (int c = 0; c < C; ++c) {
            float xi = xs[i][c];
            hp[q * 2 * C + c]     = (_Float16)xi;
            hp[q * 2 * C + C + c] = (_Float16)(xs[j][c] - xi);
        }
    }
}

// ---------------------------------------------------------------------------
// Generic WMMA GEMM + fused bias/ReLU/BN epilogue.
//   Out[M,N] = f16( gam*(relu(A@W + b) - mean)*rsqrt(var+eps) + bet )
// A: [M,K] f16 row-major.  Bp: packed f16 (see pack_w_kernel).
// Block = 128 threads = 4 waves; block tile 128x64; wave tile 32x64
// (two 16-row A fragments share each B fragment -> 8 WMMA / 12 b128 loads).
// M % 128 == 0, N % 64 == 0, K % 32 == 0 (true for every layer here).
// ---------------------------------------------------------------------------
__global__ __launch_bounds__(128)
void gemm_bn_relu_kernel(const _Float16* __restrict__ A,
                         const _Float16* __restrict__ Bp,
                         const float* __restrict__ bias,
                         const float* __restrict__ gam,
                         const float* __restrict__ bet,
                         const float* __restrict__ mean,
                         const float* __restrict__ var,
                         _Float16* __restrict__ Out,
                         int M, int N, int K) {
    const int lane = threadIdx.x & 31;
    const int wave = threadIdx.x >> 5;
    const int half = lane >> 4;        // 0: lanes 0-15, 1: lanes 16-31
    const int l16  = lane & 15;
    const int row0 = blockIdx.x * 128 + wave * 32;
    const int n0   = blockIdx.y * 64;

    // A fragment base: lanes 0-15 take K chunks {+0,+16}, lanes 16-31 {+8,+24}
    const _Float16* arow0 = A + (size_t)(row0 + l16) * K + half * 8;
    const _Float16* arow1 = arow0 + (size_t)16 * K;

    v8f acc[2][4];
#pragma unroll
    for (int f = 0; f < 2; ++f)
#pragma unroll
        for (int nt = 0; nt < 4; ++nt) acc[f][nt] = (v8f){};

    const int ksteps = K >> 5;
    for (int kt = 0; kt < ksteps; ++kt) {
        v16h a0 = load_frag16(arow0 + kt * 32, arow0 + kt * 32 + 16);
        v16h a1 = load_frag16(arow1 + kt * 32, arow1 + kt * 32 + 16);
#pragma unroll
        for (int nt = 0; nt < 4; ++nt) {
            const _Float16* bb =
                Bp + (((size_t)kt * N + (n0 + nt * 16 + l16)) << 5) + (half << 3);
            v16h bf = load_frag16(bb, bb + 16);
            acc[0][nt] = __builtin_amdgcn_wmma_f32_16x16x32_f16(
                false, a0, false, bf, (short)0, acc[0][nt], false, false);
            acc[1][nt] = __builtin_amdgcn_wmma_f32_16x16x32_f16(
                false, a1, false, bf, (short)0, acc[1][nt], false, false);
        }
    }

    // Epilogue: z = acc + b; y = sc*relu(z) + sh,  sc = g*rsqrt(v+eps),
    // sh = t - m*sc.  C/D layout: VGPR r -> (M = r + 8*half, N = l16).
#pragma unroll
    for (int nt = 0; nt < 4; ++nt) {
        int n = n0 + nt * 16 + l16;
        float b  = bias[n];
        float sc = gam[n] * rsqrtf(var[n] + BN_EPS);
        float sh = bet[n] - mean[n] * sc;
#pragma unroll
        for (int f = 0; f < 2; ++f) {
#pragma unroll
            for (int r = 0; r < 8; ++r) {
                int m = row0 + f * 16 + r + half * 8;
                float z = acc[f][nt][r] + b;
                z = z > 0.f ? z : 0.f;
                Out[(size_t)m * N + n] = (_Float16)(sc * z + sh);
            }
        }
    }
}

// ---------------------------------------------------------------------------
// Max over k=5 neighbor rows: [P*5, C] -> [P, C] (f16, optional f32 copy)
// ---------------------------------------------------------------------------
__global__ void maxk_kernel(const _Float16* __restrict__ g,
                            _Float16* __restrict__ xh,
                            float* __restrict__ xf, int P, int C) {
    size_t t = (size_t)blockIdx.x * blockDim.x + threadIdx.x;
    if (t >= (size_t)P * C) return;
    int p = (int)(t / C), c = (int)(t - (size_t)p * C);
    const _Float16* gp = g + ((size_t)p * 5) * C + c;
    float m = (float)gp[0];
#pragma unroll
    for (int q = 1; q < 5; ++q) {
        float v = (float)gp[(size_t)q * C];
        m = v > m ? v : m;
    }
    xh[t] = (_Float16)m;
    if (xf) xf[t] = m;
}

// ---------------------------------------------------------------------------
// concat [x1 | x2] -> feat [P, 192]
// ---------------------------------------------------------------------------
__global__ void concat_kernel(const _Float16* __restrict__ x1,
                              const _Float16* __restrict__ x2,
                              _Float16* __restrict__ feat, int P) {
    int p = blockIdx.x * blockDim.x + threadIdx.x;
    if (p >= P) return;
    _Float16* fp = feat + (size_t)p * 192;
    const _Float16* a = x1 + (size_t)p * 64;
    const _Float16* b = x2 + (size_t)p * 128;
#pragma unroll
    for (int c = 0; c < 64; ++c) fp[c] = a[c];
#pragma unroll
    for (int c = 0; c < 128; ++c) fp[64 + c] = b[c];
}

// ---------------------------------------------------------------------------
// Global max pool over N=32 points per sample: [B*32, C] -> [B, C]
// ---------------------------------------------------------------------------
__global__ void maxn_kernel(const _Float16* __restrict__ h,
                            _Float16* __restrict__ p, int B, int C) {
    size_t t = (size_t)blockIdx.x * blockDim.x + threadIdx.x;
    if (t >= (size_t)B * C) return;
    int s = (int)(t / C), c = (int)(t - (size_t)s * C);
    const _Float16* hp = h + ((size_t)s * 32) * C + c;
    float m = (float)hp[0];
#pragma unroll 1
    for (int q = 1; q < 32; ++q) {
        float v = (float)hp[(size_t)q * C];
        m = v > m ? v : m;
    }
    p[t] = (_Float16)m;
}

// ---------------------------------------------------------------------------
// Final projection [M,256] @ [256,2] + b -> f32 out (N too small for WMMA)
// ---------------------------------------------------------------------------
__global__ void final_kernel(const _Float16* __restrict__ A,
                             const float* __restrict__ W,
                             const float* __restrict__ b,
                             float* __restrict__ out, int M, int K, int N) {
    int t = blockIdx.x * blockDim.x + threadIdx.x;
    if (t >= M * N) return;
    int r = t / N, n = t - r * N;
    float acc = b[n];
    const _Float16* ap = A + (size_t)r * K;
    for (int k = 0; k < K; ++k) acc += (float)ap[k] * W[k * N + n];
    out[t] = acc;
}

// ---------------------------------------------------------------------------
// Host-side orchestration
// ---------------------------------------------------------------------------
static inline int cdiv(long long a, int b) { return (int)((a + b - 1) / b); }

extern "C" void kernel_launch(void* const* d_in, const int* in_sizes, int n_in,
                              void* d_out, int out_size, void* d_ws, size_t ws_size,
                              hipStream_t stream) {
    (void)in_sizes; (void)n_in; (void)out_size; (void)ws_size;

    // ---- inputs (setup_inputs order) ----
    const float* pos  = (const float*)d_in[0];
    const float* w0   = (const float*)d_in[2];
    const float* b0   = (const float*)d_in[3];
    const float* e1w0 = (const float*)d_in[4];  const float* e1b0 = (const float*)d_in[5];
    const float* e1g0 = (const float*)d_in[6];  const float* e1t0 = (const float*)d_in[7];
    const float* e1m0 = (const float*)d_in[8];  const float* e1v0 = (const float*)d_in[9];
    const float* e1w1 = (const float*)d_in[10]; const float* e1b1 = (const float*)d_in[11];
    const float* e1g1 = (const float*)d_in[12]; const float* e1t1 = (const float*)d_in[13];
    const float* e1m1 = (const float*)d_in[14]; const float* e1v1 = (const float*)d_in[15];
    const float* e1w2 = (const float*)d_in[16]; const float* e1b2 = (const float*)d_in[17];
    const float* e1g2 = (const float*)d_in[18]; const float* e1t2 = (const float*)d_in[19];
    const float* e1m2 = (const float*)d_in[20]; const float* e1v2 = (const float*)d_in[21];
    const float* e2w0 = (const float*)d_in[22]; const float* e2b0 = (const float*)d_in[23];
    const float* e2g0 = (const float*)d_in[24]; const float* e2t0 = (const float*)d_in[25];
    const float* e2m0 = (const float*)d_in[26]; const float* e2v0 = (const float*)d_in[27];
    const float* l1w  = (const float*)d_in[28]; const float* l1b  = (const float*)d_in[29];
    const float* l1g  = (const float*)d_in[30]; const float* l1t  = (const float*)d_in[31];
    const float* l1m  = (const float*)d_in[32]; const float* l1v  = (const float*)d_in[33];
    const float* hw0  = (const float*)d_in[34]; const float* hb0  = (const float*)d_in[35];
    const float* hg0  = (const float*)d_in[36]; const float* ht0  = (const float*)d_in[37];
    const float* hm0  = (const float*)d_in[38]; const float* hv0  = (const float*)d_in[39];
    const float* hw1  = (const float*)d_in[40]; const float* hb1  = (const float*)d_in[41];
    const float* hg1  = (const float*)d_in[42]; const float* ht1  = (const float*)d_in[43];
    const float* hm1  = (const float*)d_in[44]; const float* hv1  = (const float*)d_in[45];
    const float* hw2  = (const float*)d_in[46]; const float* hb2  = (const float*)d_in[47];
    float* out = (float*)d_out;

    // ---- problem constants ----
    const int B = 2048, N = 32, P = B * N;      // 65536 points
    const int R5 = P * 5;                        // 327680 edge rows

    // ---- workspace layout (bytes), with region aliasing; peak ~228 MB ----
    char* ws = (char*)d_ws;
    float*    XF   = (float*)   (ws + 0);                    //  8.0 MB [P,32] f32
    _Float16* PK   = (_Float16*)(ws + 8388608);              //  2.0 MB packed weights
    _Float16* H1   = (_Float16*)(ws + 10485760);             // 40.0 MB [R5,64]
    _Float16* G1   = (_Float16*)(ws + 52428800);             // 40.0 MB [R5,64]
    _Float16* H2   = (_Float16*)(ws + 10485760);             // 80.0 MB [R5,128] (aliases H1+G1)
    _Float16* G2   = (_Float16*)(ws + 94371840);             // 80.0 MB [R5,128]
    _Float16* X1   = (_Float16*)(ws + 178257920);            //  8.0 MB [P,64]
    _Float16* X2   = (_Float16*)(ws + 186646528);            // 16.0 MB [P,128]
    _Float16* FEAT = (_Float16*)(ws + 10485760);             // 24.0 MB [P,192] (aliases H2)
    _Float16* HB   = (_Float16*)(ws + 35651584);             //128.0 MB [P,1024]
    _Float16* PP   = (_Float16*)(ws + 203423744);            //  4.0 MB [B,1024]
    _Float16* Q0   = (_Float16*)(ws + 207618048);            //  2.0 MB [B,512]
    _Float16* Q1   = (_Float16*)(ws + 209715200);            //  1.0 MB [B,256]
    float*    X1F  = (float*)   (ws + 210763776);            // 16.0 MB [P,64] f32

    // packed weight sub-offsets (halves)
    _Float16* pE10 = PK;                 // 64x64
    _Float16* pE11 = PK + 4096;
    _Float16* pE12 = PK + 8192;
    _Float16* pE2  = PK + 12288;         // 128x128
    _Float16* pL1  = PK + 28672;         // 192x1024
    _Float16* pH0  = PK + 225280;        // 1024x512
    _Float16* pH1  = PK + 749568;        // 512x256

    // ---- 1. pack all WMMA weights ----
    pack_w_kernel<<<cdiv(64 * 64, 256), 256, 0, stream>>>(e1w0, pE10, 64, 64);
    pack_w_kernel<<<cdiv(64 * 64, 256), 256, 0, stream>>>(e1w1, pE11, 64, 64);
    pack_w_kernel<<<cdiv(64 * 64, 256), 256, 0, stream>>>(e1w2, pE12, 64, 64);
    pack_w_kernel<<<cdiv(128 * 128, 256), 256, 0, stream>>>(e2w0, pE2, 128, 128);
    pack_w_kernel<<<cdiv(192 * 1024, 256), 256, 0, stream>>>(l1w, pL1, 192, 1024);
    pack_w_kernel<<<cdiv(1024 * 512, 256), 256, 0, stream>>>(hw0, pH0, 1024, 512);
    pack_w_kernel<<<cdiv(512 * 256, 256), 256, 0, stream>>>(hw1, pH1, 512, 256);

    // ---- 2. conv0 ----
    conv0_kernel<<<cdiv(P, 256), 256, 0, stream>>>(pos, w0, b0, XF, P);

    // ---- 3. EdgeConv1: kNN + edge features, 3x (WMMA GEMM + ReLU + BN), max_k ----
    knn_edge_kernel<32><<<B, 32, 0, stream>>>(XF, H1);
    {
        dim3 g(R5 / 128, 1);
        gemm_bn_relu_kernel<<<g, 128, 0, stream>>>(H1, pE10, e1b0, e1g0, e1t0, e1m0, e1v0, G1, R5, 64, 64);
        gemm_bn_relu_kernel<<<g, 128, 0, stream>>>(G1, pE11, e1b1, e1g1, e1t1, e1m1, e1v1, H1, R5, 64, 64);
        gemm_bn_relu_kernel<<<g, 128, 0, stream>>>(H1, pE12, e1b2, e1g2, e1t2, e1m2, e1v2, G1, R5, 64, 64);
    }
    maxk_kernel<<<cdiv((long long)P * 64, 256), 256, 0, stream>>>(G1, X1, X1F, P, 64);

    // ---- 4. EdgeConv2 ----
    knn_edge_kernel<64><<<B, 32, 0, stream>>>(X1F, H2);
    {
        dim3 g(R5 / 128, 2);
        gemm_bn_relu_kernel<<<g, 128, 0, stream>>>(H2, pE2, e2b0, e2g0, e2t0, e2m0, e2v0, G2, R5, 128, 128);
    }
    maxk_kernel<<<cdiv((long long)P * 128, 256), 256, 0, stream>>>(G2, X2, (float*)nullptr, P, 128);

    // ---- 5. concat + l1 (192->1024) + global max pool ----
    concat_kernel<<<cdiv(P, 256), 256, 0, stream>>>(X1, X2, FEAT, P);
    {
        dim3 g(P / 128, 1024 / 64);
        gemm_bn_relu_kernel<<<g, 128, 0, stream>>>(FEAT, pL1, l1b, l1g, l1t, l1m, l1v, HB, P, 1024, 192);
    }
    maxn_kernel<<<cdiv((long long)B * 1024, 256), 256, 0, stream>>>(HB, PP, B, 1024);

    // ---- 6. head MLP ----
    {
        dim3 g(B / 128, 512 / 64);
        gemm_bn_relu_kernel<<<g, 128, 0, stream>>>(PP, pH0, hb0, hg0, ht0, hm0, hv0, Q0, B, 512, 1024);
    }
    {
        dim3 g(B / 128, 256 / 64);
        gemm_bn_relu_kernel<<<g, 128, 0, stream>>>(Q0, pH1, hb1, hg1, ht1, hm1, hv1, Q1, B, 256, 512);
    }
    final_kernel<<<cdiv(B * 2, 256), 256, 0, stream>>>(Q1, hw2, hb2, out, B, 256, 2);
}
